// WeightDropLstm_60095182406267
// MI455X (gfx1250) — matrix operand order repaired and checked by reading, not compiled, not run on hardware
//
#include <hip/hip_runtime.h>

// ---------------- problem constants ----------------
constexpr int B_ = 4096, T_ = 70, E_ = 300, H_ = 60;
constexpr int BT = B_ * T_;             // 286720
constexpr int EP = 320;                 // E padded to mult of 32
constexpr int LSTM_N = 512;             // 4H*2 dirs = 480, padded to 512 (64-wide N tiles)
constexpr int GRU_N = 384;              // 3H=180 per dir, each padded to 192
constexpr int GRU_K = 128;              // 2H=120 padded
constexpr int HK = 64;                  // H padded for recurrent K

typedef __attribute__((ext_vector_type(16))) __bf16 bf16x16;
typedef __attribute__((ext_vector_type(8)))  __bf16 bf16x8;
typedef __attribute__((ext_vector_type(8)))  float  f32x8;

#define WMMA_BF16(a, b, c) \
  __builtin_amdgcn_wmma_f32_16x16x32_bf16(false, (a), false, (b), (short)0, (c), false, false)

#define WAIT_ASYNC(imm) asm volatile("s_wait_asynccnt " #imm ::: "memory")

// Async DMA: global memory -> LDS, 16 bytes per lane, tracked by ASYNCcnt.
__device__ __forceinline__ void async_ld_b128(uint32_t lds_off, uint64_t gaddr) {
  asm volatile("global_load_async_to_lds_b128 %0, %1, off"
               :: "v"(lds_off), "v"(gaddr) : "memory");
}
__device__ __forceinline__ uint32_t lds_addr(const void* p) {
  return (uint32_t)(size_t)p;  // LDS aperture: addr[31:0] is the LDS byte offset
}

// A fragment: 16x32 tile, row-major in LDS, lane = M, half-lane interleave of K 8-blocks
__device__ __forceinline__ bf16x16 frag_a(const __bf16* p, int stride) {
  int lane = threadIdx.x & 31;
  int m = lane & 15, h = lane >> 4;
  const __bf16* r0 = p + m * stride + h * 8;
  bf16x8 lo = *(const bf16x8*)(r0);
  bf16x8 hi = *(const bf16x8*)(r0 + 16);
  bf16x16 out;
#pragma unroll
  for (int i = 0; i < 8; i++) { out[i] = lo[i]; out[i + 8] = hi[i]; }
  return out;
}

// B fragment: 32x16 (KxN) from row-major W[N][K] tile in LDS, lane = N, half-lane interleave of K 8-blocks
__device__ __forceinline__ bf16x16 frag_b(const __bf16* p, int stride) {
  int lane = threadIdx.x & 31;
  int n = lane & 15, h = lane >> 4;
  const __bf16* r0 = p + n * stride + h * 8;
  bf16x8 lo = *(const bf16x8*)(r0);
  bf16x8 hi = *(const bf16x8*)(r0 + 16);
  bf16x16 out;
#pragma unroll
  for (int i = 0; i < 8; i++) { out[i] = lo[i]; out[i + 8] = hi[i]; }
  return out;
}

__device__ __forceinline__ float sigmoidf_(float x) { return 1.0f / (1.0f + expf(-x)); }

// ---------------- small prep kernels ----------------
__global__ void zero_bf16_k(__bf16* p, int n) {
  int i = blockIdx.x * 256 + threadIdx.x;
  if (i < n) p[i] = (__bf16)0.0f;
}
__global__ void zero_f32_k(float* p, int n) {
  int i = blockIdx.x * 256 + threadIdx.x;
  if (i < n) p[i] = 0.0f;
}
__global__ void cast_copy_k(__bf16* dst, int dstStride, int rowOff,
                            const float* src, int rows, int cols) {
  int i = blockIdx.x * 256 + threadIdx.x;
  if (i < rows * cols) {
    int r = i / cols, c = i - r * cols;
    dst[(size_t)(rowOff + r) * dstStride + c] = (__bf16)src[i];
  }
}
__global__ void copy_f32_k(float* dst, const float* src, int n) {
  int i = blockIdx.x * 256 + threadIdx.x;
  if (i < n) dst[i] = src[i];
}

// ---------------- embedding gather + bf16 cast ----------------
__global__ __launch_bounds__(128) void embed_cast_k(const int* tokens, const float* emb,
                                                    __bf16* x) {
  size_t bt = blockIdx.x;
  int tok = tokens[bt];
  const float* src = emb + (size_t)tok * E_;
  __bf16* dst = x + bt * EP;
  for (int e = threadIdx.x; e < EP; e += 128)
    dst[e] = (e < E_) ? (__bf16)src[e] : (__bf16)0.0f;
}

// ------- WMMA GEMM, 64x64 tile, double-buffered async-to-LDS pipeline ----------
// C[M][ldc] = A[M][K](bf16) @ W[N][K]^T + bias
__global__ __launch_bounds__(256) void gemm_bias_k(const __bf16* __restrict__ A,
                                                   const __bf16* __restrict__ W,
                                                   const float* __restrict__ bias,
                                                   float* __restrict__ C, int K, int ldc) {
  __shared__ __align__(16) __bf16 sA[2][64 * 32];
  __shared__ __align__(16) __bf16 sB[2][64 * 32];
  int tid = threadIdx.x;
  int wave = tid >> 5, lane = tid & 31;
  int m0 = blockIdx.y * 64;
  int n0 = blockIdx.x * 64;
  int mrow = wave & 3;          // 0..3
  int nc0 = (wave >> 2) * 2;    // 0 or 2: this wave's two N tiles
  int r = tid >> 2, ch = tid & 3;  // per-thread 16B chunk (row r, chunk ch) for A and B
  const __bf16* Ap = A + (size_t)(m0 + r) * K + ch * 8;
  const __bf16* Wp = W + (size_t)(n0 + r) * K + ch * 8;
  uint32_t ldsA0 = lds_addr(&sA[0][r * 32 + ch * 8]);
  uint32_t ldsA1 = lds_addr(&sA[1][r * 32 + ch * 8]);
  uint32_t ldsB0 = lds_addr(&sB[0][r * 32 + ch * 8]);
  uint32_t ldsB1 = lds_addr(&sB[1][r * 32 + ch * 8]);
  f32x8 acc0 = {}, acc1 = {};
  int nk = K >> 5;
  // prologue: stage 0
  async_ld_b128(ldsA0, (uint64_t)(size_t)Ap);
  async_ld_b128(ldsB0, (uint64_t)(size_t)Wp);
  for (int kk = 0; kk < nk; kk++) {
    int cur = kk & 1;
    if (kk + 1 < nk) {  // issue stage kk+1 into the other buffer, then retire stage kk
      async_ld_b128(cur ? ldsA0 : ldsA1, (uint64_t)(size_t)(Ap + (kk + 1) * 32));
      async_ld_b128(cur ? ldsB0 : ldsB1, (uint64_t)(size_t)(Wp + (kk + 1) * 32));
      WAIT_ASYNC(0x2);
    } else {
      WAIT_ASYNC(0x0);
    }
    __syncthreads();
    bf16x16 fa = frag_a(&sA[cur][mrow * 16 * 32], 32);
    bf16x16 fb0 = frag_b(&sB[cur][nc0 * 16 * 32], 32);
    bf16x16 fb1 = frag_b(&sB[cur][(nc0 + 1) * 16 * 32], 32);
    acc0 = WMMA_BF16(fa, fb0, acc0);
    acc1 = WMMA_BF16(fa, fb1, acc1);
    __syncthreads();
  }
  int nA = n0 + nc0 * 16 + (lane & 15);
  float bs0 = bias[nA];
  float bs1 = bias[nA + 16];
#pragma unroll
  for (int rr = 0; rr < 8; rr++) {
    int m = m0 + mrow * 16 + ((lane < 16) ? rr : rr + 8);
    C[(size_t)m * ldc + nA] = acc0[rr] + bs0;
    C[(size_t)m * ldc + nA + 16] = acc1[rr] + bs1;
  }
}

// ---------------- persistent LSTM cell: 16 batch rows / block, 70 steps ----------------
// gi rows are DMA'd one step ahead into a double-buffered LDS stage.
__global__ __launch_bounds__(128) void lstm_cell_k(const float* __restrict__ gi, int giOff,
                                                   const __bf16* __restrict__ Whh,  // [240][64]
                                                   const float* __restrict__ bhh,   // [240]
                                                   __bf16* __restrict__ y,          // [BT][128]
                                                   int yOff, int reverse) {
  __shared__ __align__(16) __bf16 sW[240 * 64];
  __shared__ __align__(16) __bf16 sH[16 * 64];
  __shared__ float sC[16 * 60];
  __shared__ float sG[16 * 240];
  __shared__ __align__(16) float sGI[2][16 * 256];  // 256 floats/row staged (240 used)
  int tid = threadIdx.x, wave = tid >> 5, lane = tid & 31;
  int brow0 = blockIdx.x * 16;
  int row8 = tid >> 3, j8 = tid & 7;  // async mapping: 16 rows x 64 chunks, 8 chunks/thread
  {
    const uint32_t* src = (const uint32_t*)Whh;
    uint32_t* dst = (uint32_t*)sW;
    for (int i = tid; i < 240 * 32; i += 128) dst[i] = src[i];
  }
  for (int i = tid; i < 16 * 64; i += 128) sH[i] = (__bf16)0.0f;
  for (int i = tid; i < 16 * 60; i += 128) sC[i] = 0.0f;
  size_t rowbase = ((size_t)(brow0 + row8) * T_) * LSTM_N + giOff;  // + t*LSTM_N at use

  // prologue: stage for first step
  {
    int t0 = reverse ? (T_ - 1) : 0;
    const float* src = gi + rowbase + (size_t)t0 * LSTM_N;
#pragma unroll
    for (int k = 0; k < 8; k++)
      async_ld_b128(lds_addr(&sGI[0][row8 * 256 + (j8 + k * 8) * 4]),
                    (uint64_t)(size_t)(src + (j8 + k * 8) * 4));
  }
  __syncthreads();
  for (int s = 0; s < T_; s++) {
    int t = reverse ? (T_ - 1 - s) : s;
    if (s + 1 < T_) {
      int tn = reverse ? (T_ - 2 - s) : (s + 1);
      const float* src = gi + rowbase + (size_t)tn * LSTM_N;
      int buf = (s + 1) & 1;
#pragma unroll
      for (int k = 0; k < 8; k++)
        async_ld_b128(lds_addr(&sGI[buf][row8 * 256 + (j8 + k * 8) * 4]),
                      (uint64_t)(size_t)(src + (j8 + k * 8) * 4));
      WAIT_ASYNC(0x8);  // retire stage s (8 older async ops), stage s+1 in flight
    } else {
      WAIT_ASYNC(0x0);
    }
    __syncthreads();  // stage-s gi visible to all waves; sH update from prev step visible
    const float* gstage = sGI[s & 1];
    bf16x16 a0 = frag_a(sH, 64);
    bf16x16 a1 = frag_a(sH + 32, 64);
    for (int nt = wave; nt < 15; nt += 4) {
      f32x8 acc = {};
      bf16x16 w0 = frag_b(&sW[nt * 16 * 64], 64);
      bf16x16 w1 = frag_b(&sW[nt * 16 * 64 + 32], 64);
      acc = WMMA_BF16(a0, w0, acc);
      acc = WMMA_BF16(a1, w1, acc);
      int n = nt * 16 + (lane & 15);
      float bb = bhh[n];
#pragma unroll
      for (int rr = 0; rr < 8; rr++) {
        int m = (lane < 16) ? rr : rr + 8;
        sG[m * 240 + n] = acc[rr] + gstage[m * 256 + n] + bb;
      }
    }
    __syncthreads();
    for (int i = tid; i < 960; i += 128) {
      int m = i / 60, n = i - m * 60;
      float xi = sG[m * 240 + n];
      float xf = sG[m * 240 + 60 + n];
      float xg = sG[m * 240 + 120 + n];
      float xo = sG[m * 240 + 180 + n];
      float c = sigmoidf_(xf) * sC[i] + sigmoidf_(xi) * tanhf(xg);
      float h = sigmoidf_(xo) * tanhf(c);
      sC[i] = c;
      sH[m * 64 + n] = (__bf16)h;
      y[((size_t)(brow0 + m) * T_ + t) * GRU_K + yOff + n] = (__bf16)h;
    }
  }
}

// ---------------- persistent GRU cell ----------------
__global__ __launch_bounds__(128) void gru_cell_k(const float* __restrict__ gi, int giOff,
                                                  const __bf16* __restrict__ Whh,  // [192][64]
                                                  const float* __restrict__ bhh,   // [192]
                                                  float* __restrict__ hout,        // [BT][120]
                                                  int hOff, float* __restrict__ ghT,  // [B][60]
                                                  int reverse) {
  __shared__ __align__(16) __bf16 sW[192 * 64];
  __shared__ __align__(16) __bf16 sH[16 * 64];
  __shared__ float sHf[16 * 60];
  __shared__ float sG[16 * 192];
  __shared__ __align__(16) float sGI[2][16 * 192];  // 192 floats/row = 48 chunks/row
  int tid = threadIdx.x, wave = tid >> 5, lane = tid & 31;
  int brow0 = blockIdx.x * 16;
  int row8 = tid >> 3, j8 = tid & 7;  // 16 rows x 48 chunks, 6 chunks/thread
  {
    const uint32_t* src = (const uint32_t*)Whh;
    uint32_t* dst = (uint32_t*)sW;
    for (int i = tid; i < 192 * 32; i += 128) dst[i] = src[i];
  }
  for (int i = tid; i < 16 * 64; i += 128) sH[i] = (__bf16)0.0f;
  for (int i = tid; i < 16 * 60; i += 128) sHf[i] = 0.0f;
  size_t rowbase = ((size_t)(brow0 + row8) * T_) * GRU_N + giOff;

  {
    int t0 = reverse ? (T_ - 1) : 0;
    const float* src = gi + rowbase + (size_t)t0 * GRU_N;
#pragma unroll
    for (int k = 0; k < 6; k++)
      async_ld_b128(lds_addr(&sGI[0][row8 * 192 + (j8 + k * 8) * 4]),
                    (uint64_t)(size_t)(src + (j8 + k * 8) * 4));
  }
  __syncthreads();
  for (int s = 0; s < T_; s++) {
    int t = reverse ? (T_ - 1 - s) : s;
    if (s + 1 < T_) {
      int tn = reverse ? (T_ - 2 - s) : (s + 1);
      const float* src = gi + rowbase + (size_t)tn * GRU_N;
      int buf = (s + 1) & 1;
#pragma unroll
      for (int k = 0; k < 6; k++)
        async_ld_b128(lds_addr(&sGI[buf][row8 * 192 + (j8 + k * 8) * 4]),
                      (uint64_t)(size_t)(src + (j8 + k * 8) * 4));
      WAIT_ASYNC(0x6);
    } else {
      WAIT_ASYNC(0x0);
    }
    __syncthreads();
    const float* gstage = sGI[s & 1];
    bf16x16 a0 = frag_a(sH, 64);
    bf16x16 a1 = frag_a(sH + 32, 64);
    for (int nt = wave; nt < 12; nt += 4) {
      f32x8 acc = {};
      bf16x16 w0 = frag_b(&sW[nt * 16 * 64], 64);
      bf16x16 w1 = frag_b(&sW[nt * 16 * 64 + 32], 64);
      acc = WMMA_BF16(a0, w0, acc);
      acc = WMMA_BF16(a1, w1, acc);
      int n = nt * 16 + (lane & 15);
      float bb = bhh[n];
#pragma unroll
      for (int rr = 0; rr < 8; rr++) {
        int m = (lane < 16) ? rr : rr + 8;
        sG[m * 192 + n] = acc[rr] + bb;
      }
    }
    __syncthreads();
    for (int i = tid; i < 960; i += 128) {
      int m = i / 60, n = i - m * 60;
      float r_ = sigmoidf_(gstage[m * 192 + n] + sG[m * 192 + n]);
      float z = sigmoidf_(gstage[m * 192 + 60 + n] + sG[m * 192 + 60 + n]);
      float nn = tanhf(gstage[m * 192 + 120 + n] + r_ * sG[m * 192 + 120 + n]);
      float h = (1.0f - z) * nn + z * sHf[i];
      sHf[i] = h;
      sH[m * 64 + n] = (__bf16)h;
      hout[((size_t)(brow0 + m) * T_ + t) * 120 + hOff + n] = h;
    }
  }
  __syncthreads();
  for (int i = tid; i < 960; i += 128) {
    int m = i / 60, n = i - m * 60;
    ghT[(size_t)(brow0 + m) * 60 + n] = sHf[i];
  }
}

// ---------------- attention + pooling + MLP head (1 block per batch row) ------------
__global__ __launch_bounds__(128) void head_k(
    const float* __restrict__ hgru, const float* __restrict__ ghf,
    const float* __restrict__ ghb, const float* __restrict__ fin,
    const float* __restrict__ attn_w, const float* __restrict__ attn_b,
    const float* __restrict__ linW, const float* __restrict__ linb,
    const float* __restrict__ gamma, const float* __restrict__ beta,
    const float* __restrict__ mean, const float* __restrict__ var,
    const float* __restrict__ outW, const float* __restrict__ outb,
    float* __restrict__ out) {
  int b = blockIdx.x, tid = threadIdx.x;
  __shared__ float sA[T_];
  __shared__ float sConc[481];
  __shared__ float sInv;
  __shared__ float sL[16];
  const float* hb = hgru + (size_t)b * T_ * 120;
  for (int t = tid; t < T_; t += 128) {
    float d = 0.0f;
    for (int j = 0; j < 120; j++) d += hb[t * 120 + j] * attn_w[j];
    sA[t] = expf(tanhf(d) + attn_b[t]);
  }
  __syncthreads();
  if (tid == 0) {
    float s = 0.0f;
    for (int t = 0; t < T_; t++) s += sA[t];
    sInv = 1.0f / (s + 1e-10f);
  }
  __syncthreads();
  float inv = sInv;
  for (int j = tid; j < 120; j += 128) {
    float att = 0.0f, av = 0.0f, mx = -3.0e38f;
    for (int t = 0; t < T_; t++) {
      float v = hb[t * 120 + j];
      att += v * sA[t];
      av += v;
      mx = fmaxf(mx, v);
    }
    sConc[120 + j] = att * inv;
    sConc[240 + j] = av * (1.0f / (float)T_);
    sConc[360 + j] = mx;
  }
  for (int j = tid; j < 120; j += 128) {
    int half = (j >= 60) ? 1 : 0;
    int jj = j - 60 * half;
    float v;
    if (b < B_ / 2) v = ghf[(size_t)(2 * b + half) * 60 + jj];
    else            v = ghb[(size_t)(2 * (b - B_ / 2) + half) * 60 + jj];
    sConc[j] = v;
  }
  if (tid == 0) sConc[480] = fin[b];
  __syncthreads();
  if (tid < 16) {
    float o = linb[tid];
    for (int k = 0; k < 481; k++) o += sConc[k] * linW[tid * 481 + k];
    o = fmaxf(o, 0.0f);
    o = (o - mean[tid]) * rsqrtf(var[tid] + 1e-5f) * gamma[tid] + beta[tid];
    sL[tid] = o;
  }
  __syncthreads();
  if (tid == 0) {
    float o = outb[0];
    for (int k = 0; k < 16; k++) o += sL[k] * outW[k];
    out[b] = o;
  }
}

// ---------------- launcher ----------------
extern "C" void kernel_launch(void* const* d_in, const int* in_sizes, int n_in,
                              void* d_out, int out_size, void* d_ws, size_t ws_size,
                              hipStream_t stream) {
  (void)in_sizes; (void)n_in; (void)out_size; (void)ws_size;
  const int*   tokens    = (const int*)d_in[0];
  const float* fin       = (const float*)d_in[1];
  const float* emb       = (const float*)d_in[2];
  const float* lWih_f    = (const float*)d_in[3];
  const float* lWhh_f    = (const float*)d_in[4];
  const float* lbih_f    = (const float*)d_in[5];
  const float* lbhh_f    = (const float*)d_in[6];
  const float* lWih_b    = (const float*)d_in[7];
  const float* lWhh_b    = (const float*)d_in[8];
  const float* lbih_b    = (const float*)d_in[9];
  const float* lbhh_b    = (const float*)d_in[10];
  const float* gWih_f    = (const float*)d_in[11];
  const float* gWhh_f    = (const float*)d_in[12];
  const float* gbih_f    = (const float*)d_in[13];
  const float* gbhh_f    = (const float*)d_in[14];
  const float* gWih_b    = (const float*)d_in[15];
  const float* gWhh_b    = (const float*)d_in[16];
  const float* gbih_b    = (const float*)d_in[17];
  const float* gbhh_b    = (const float*)d_in[18];

  char* ws = (char*)d_ws;
  size_t off = 0;
  auto take = [&](size_t bytes) { size_t r = off; off = (off + bytes + 255) & ~(size_t)255; return r; };
  __bf16* x_bf16   = (__bf16*)(ws + take((size_t)BT * EP * 2));
  float*  gi_lstm  = (float*) (ws + take((size_t)BT * LSTM_N * 4));
  float*  gi_gru   = (float*) (ws + take((size_t)BT * GRU_N * 4));
  __bf16* h_lstm   = (__bf16*)(ws + take((size_t)BT * GRU_K * 2));
  float*  h_gru    = (float*) (ws + take((size_t)BT * 120 * 4));
  float*  ghf      = (float*) (ws + take((size_t)B_ * 60 * 4));
  float*  ghb      = (float*) (ws + take((size_t)B_ * 60 * 4));
  __bf16* wih_l    = (__bf16*)(ws + take((size_t)LSTM_N * EP * 2));
  __bf16* whh_lf   = (__bf16*)(ws + take((size_t)240 * HK * 2));
  __bf16* whh_lb   = (__bf16*)(ws + take((size_t)240 * HK * 2));
  __bf16* wih_g    = (__bf16*)(ws + take((size_t)GRU_N * GRU_K * 2));
  __bf16* whh_gf   = (__bf16*)(ws + take((size_t)192 * HK * 2));
  __bf16* whh_gb   = (__bf16*)(ws + take((size_t)192 * HK * 2));
  float*  bih_l    = (float*) (ws + take((size_t)LSTM_N * 4));
  float*  bih_g    = (float*) (ws + take((size_t)GRU_N * 4));
  float*  bhh_gfp  = (float*) (ws + take((size_t)192 * 4));
  float*  bhh_gbp  = (float*) (ws + take((size_t)192 * 4));

  auto g1 = [](int n) { return dim3((n + 255) / 256); };

  // ---- weight / bias prep (pad with zeros, cast to bf16) ----
  zero_bf16_k<<<g1(LSTM_N * EP), 256, 0, stream>>>(wih_l, LSTM_N * EP);
  zero_bf16_k<<<g1(240 * HK), 256, 0, stream>>>(whh_lf, 240 * HK);
  zero_bf16_k<<<g1(240 * HK), 256, 0, stream>>>(whh_lb, 240 * HK);
  zero_bf16_k<<<g1(GRU_N * GRU_K), 256, 0, stream>>>(wih_g, GRU_N * GRU_K);
  zero_bf16_k<<<g1(192 * HK), 256, 0, stream>>>(whh_gf, 192 * HK);
  zero_bf16_k<<<g1(192 * HK), 256, 0, stream>>>(whh_gb, 192 * HK);
  cast_copy_k<<<g1(240 * E_), 256, 0, stream>>>(wih_l, EP, 0,   lWih_f, 240, E_);
  cast_copy_k<<<g1(240 * E_), 256, 0, stream>>>(wih_l, EP, 240, lWih_b, 240, E_);
  cast_copy_k<<<g1(240 * H_), 256, 0, stream>>>(whh_lf, HK, 0, lWhh_f, 240, H_);
  cast_copy_k<<<g1(240 * H_), 256, 0, stream>>>(whh_lb, HK, 0, lWhh_b, 240, H_);
  cast_copy_k<<<g1(180 * 120), 256, 0, stream>>>(wih_g, GRU_K, 0,   gWih_f, 180, 120);
  cast_copy_k<<<g1(180 * 120), 256, 0, stream>>>(wih_g, GRU_K, 192, gWih_b, 180, 120);
  cast_copy_k<<<g1(180 * H_), 256, 0, stream>>>(whh_gf, HK, 0, gWhh_f, 180, H_);
  cast_copy_k<<<g1(180 * H_), 256, 0, stream>>>(whh_gb, HK, 0, gWhh_b, 180, H_);
  zero_f32_k<<<g1(LSTM_N), 256, 0, stream>>>(bih_l, LSTM_N);
  copy_f32_k<<<g1(240), 256, 0, stream>>>(bih_l,       lbih_f, 240);
  copy_f32_k<<<g1(240), 256, 0, stream>>>(bih_l + 240, lbih_b, 240);
  zero_f32_k<<<g1(GRU_N), 256, 0, stream>>>(bih_g, GRU_N);
  copy_f32_k<<<g1(180), 256, 0, stream>>>(bih_g,       gbih_f, 180);
  copy_f32_k<<<g1(180), 256, 0, stream>>>(bih_g + 192, gbih_b, 180);
  zero_f32_k<<<g1(192), 256, 0, stream>>>(bhh_gfp, 192);
  zero_f32_k<<<g1(192), 256, 0, stream>>>(bhh_gbp, 192);
  copy_f32_k<<<g1(180), 256, 0, stream>>>(bhh_gfp, gbhh_f, 180);
  copy_f32_k<<<g1(180), 256, 0, stream>>>(bhh_gbp, gbhh_b, 180);

  // ---- embedding gather + cast ----
  embed_cast_k<<<BT, 128, 0, stream>>>(tokens, emb, x_bf16);

  // ---- LSTM input projection: (BT x 320) @ (512 x 320)^T ----
  gemm_bias_k<<<dim3(LSTM_N / 64, BT / 64), 256, 0, stream>>>(x_bf16, wih_l, bih_l,
                                                              gi_lstm, EP, LSTM_N);
  // ---- LSTM recurrence (persistent, 16 rows/block) ----
  lstm_cell_k<<<B_ / 16, 128, 0, stream>>>(gi_lstm, 0,   whh_lf, lbhh_f, h_lstm, 0,  0);
  lstm_cell_k<<<B_ / 16, 128, 0, stream>>>(gi_lstm, 240, whh_lb, lbhh_b, h_lstm, 60, 1);

  // ---- GRU input projection: (BT x 128) @ (384 x 128)^T ----
  gemm_bias_k<<<dim3(GRU_N / 64, BT / 64), 256, 0, stream>>>(h_lstm, wih_g, bih_g,
                                                             gi_gru, GRU_K, GRU_N);
  // ---- GRU recurrence ----
  gru_cell_k<<<B_ / 16, 128, 0, stream>>>(gi_gru, 0,   whh_gf, bhh_gfp, h_gru, 0,  ghf, 0);
  gru_cell_k<<<B_ / 16, 128, 0, stream>>>(gi_gru, 192, whh_gb, bhh_gbp, h_gru, 60, ghb, 1);

  // ---- head ----
  head_k<<<B_, 128, 0, stream>>>(h_gru, ghf, ghb, fin,
                                 (const float*)d_in[19], (const float*)d_in[20],
                                 (const float*)d_in[21], (const float*)d_in[22],
                                 (const float*)d_in[23], (const float*)d_in[24],
                                 (const float*)d_in[25], (const float*)d_in[26],
                                 (const float*)d_in[27], (const float*)d_in[28],
                                 (float*)d_out);
}
